// MultiHeadAttention_88287347737046
// MI455X (gfx1250) — matrix operand order repaired and checked by reading, not compiled, not run on hardware
//
#include <hip/hip_runtime.h>
#include <hip/hip_bf16.h>

#define B_  256
#define T_  256
#define C_  384
#define H_  6
#define HD_ 64
#define M_  (B_ * T_)   // 65536 flattened rows

typedef __attribute__((ext_vector_type(16))) __bf16 v16bf;
typedef __attribute__((ext_vector_type(8)))  float  v8f;
typedef __attribute__((ext_vector_type(8)))  unsigned short v8us;

// ---------------------------------------------------------------- helpers ---

__device__ __forceinline__ unsigned short f2bf(float f) {
    union { float f; unsigned int u; } v;
    v.f = f;
    unsigned int u = v.u;
    u += 0x7fffu + ((u >> 16) & 1u);   // round-to-nearest-even
    return (unsigned short)(u >> 16);
}

// A-matrix 16x32 bf16 fragment: lane<16 row=lane holds K=0..15;
// lane>=16 row=lane-16 holds K=16..31 (matches load_b16).
__device__ __forceinline__ v16bf load_a16(const unsigned short* base, int ld,
                                          int row0, int k0) {
    const int lane = threadIdx.x & 31;
    const int hl = lane >> 4, r = lane & 15;
    const unsigned short* p = base + (size_t)(row0 + r) * ld + k0 + hl * 16;
    union { v16bf v; v8us u[2]; } t;
    t.u[0] = *(const v8us*)p;          // K = base..base+7
    t.u[1] = *(const v8us*)(p + 8);    // K = base+8..base+15
    return t.v;
}

// B-matrix 32x16 bf16 fragment: lane<16 col=lane holds K=0..15 (contiguous in
// a row-major [N,K] operand, e.g. W[out,in] or K[T,HD]); lane>=16 K=16..31.
__device__ __forceinline__ v16bf load_b16(const unsigned short* base, int ld,
                                          int n0, int k0) {
    const int lane = threadIdx.x & 31;
    const int hl = lane >> 4, c = lane & 15;
    const unsigned short* p = base + (size_t)(n0 + c) * ld + k0 + hl * 16;
    union { v16bf v; v8us u[2]; } t;
    t.u[0] = *(const v8us*)p;
    t.u[1] = *(const v8us*)(p + 8);
    return t.v;
}

__device__ __forceinline__ v8f wmma_bf16(v16bf a, v16bf b, v8f c) {
    return __builtin_amdgcn_wmma_f32_16x16x32_bf16(false, a, false, b,
                                                   (short)0, c, false, false);
}

// ------------------------------------------------------------- conversion ---

__global__ __launch_bounds__(256) void f32_to_bf16_kernel(
        const float* __restrict__ in, unsigned short* __restrict__ out, int n) {
    int i = blockIdx.x * blockDim.x + threadIdx.x;
    const int stride = gridDim.x * blockDim.x;
    for (; i < n; i += stride) out[i] = f2bf(in[i]);
}

// ---------------------------------------------------------------- QKV GEMM --
// Y = X @ W.T  (X bf16 [M_,C_], W bf16 [C_,C_] row-major [out,in]).
// One wave computes a 32(M) x 64(N) tile; N-tile of 64 == one head's HD slice.
// Output stored head-major [B,H,T,HD] bf16.

__global__ __launch_bounds__(32) void qkv_gemm_kernel(
        const unsigned short* __restrict__ X,
        const unsigned short* __restrict__ Wq,
        const unsigned short* __restrict__ Wk,
        const unsigned short* __restrict__ Wv,
        unsigned short* __restrict__ Qo,
        unsigned short* __restrict__ Ko,
        unsigned short* __restrict__ Vo) {
    const int m0 = blockIdx.x * 32;
    const int h  = blockIdx.y;
    const int n0 = h * HD_;
    const int w  = blockIdx.z;
    const unsigned short* W = (w == 0) ? Wq : (w == 1) ? Wk : Wv;
    unsigned short*       O = (w == 0) ? Qo : (w == 1) ? Ko : Vo;

    const v8f vzero = {0.f,0.f,0.f,0.f,0.f,0.f,0.f,0.f};
    v8f acc[2][4];
#pragma unroll
    for (int i = 0; i < 2; ++i)
#pragma unroll
        for (int j = 0; j < 4; ++j) acc[i][j] = vzero;

    for (int k0 = 0; k0 < C_; k0 += 32) {
        v16bf a0 = load_a16(X, C_, m0, k0);
        v16bf a1 = load_a16(X, C_, m0 + 16, k0);
#pragma unroll
        for (int ni = 0; ni < 4; ++ni) {
            v16bf b = load_b16(W, C_, n0 + ni * 16, k0);
            acc[0][ni] = wmma_bf16(a0, b, acc[0][ni]);
            acc[1][ni] = wmma_bf16(a1, b, acc[1][ni]);
        }
    }

    const int lane = threadIdx.x & 31;
    const int hl = lane >> 4, c = lane & 15;
#pragma unroll
    for (int mi = 0; mi < 2; ++mi)
#pragma unroll
        for (int ni = 0; ni < 4; ++ni)
#pragma unroll
            for (int r = 0; r < 8; ++r) {
                int m  = m0 + mi * 16 + r + hl * 8;   // C/D: lane<16 M=r, else r+8
                int bb = m >> 8;                       // batch (T_ == 256)
                int t  = m & 255;
                size_t o = ((size_t)(bb * H_ + h) * T_ + t) * HD_ + ni * 16 + c;
                O[o] = f2bf(acc[mi][ni][r]);
            }
}

// ------------------------------------------------------------ V transpose ---
// [B,H,T,HD] -> [B,H,HD,T] so P@V B-fragments read contiguously.

__global__ __launch_bounds__(256) void vtrans_kernel(
        const unsigned short* __restrict__ V, unsigned short* __restrict__ Vt) {
    __shared__ unsigned short tile[T_ * HD_];
    const size_t base = (size_t)blockIdx.x * T_ * HD_;
    for (int i = threadIdx.x; i < T_ * HD_; i += 256) tile[i] = V[base + i];
    __syncthreads();
    for (int i = threadIdx.x; i < T_ * HD_; i += 256) {
        int t = i & (T_ - 1);
        int d = i >> 8;                       // T_ == 256
        Vt[base + (size_t)d * T_ + t] = tile[t * HD_ + d];
    }
}

// --------------------------------------------------------------- attention --
// One wave per (b, h, 16 q-rows): S row-block in LDS, causal softmax, P@V.

__global__ __launch_bounds__(32) void attn_kernel(
        const unsigned short* __restrict__ Qh,
        const unsigned short* __restrict__ Kh,
        const unsigned short* __restrict__ Vt,
        unsigned short* __restrict__ Ob) {
    __shared__ float S[16][T_];
    __shared__ unsigned short P[16][T_];
    __shared__ float rinv[16];

    const int lane = threadIdx.x & 31;
    const int qb = blockIdx.x, h = blockIdx.y, b = blockIdx.z;
    const int q0 = qb * 16;
    const size_t headoff = (size_t)(b * H_ + h) * T_ * HD_;
    const unsigned short* Q = Qh + headoff;
    const unsigned short* K = Kh + headoff;
    const unsigned short* V = Vt + headoff;   // [HD_][T_]

    // init S to -inf-ish so unvisited key tiles vanish under softmax
    float4* S4 = (float4*)&S[0][0];
    for (int i = lane; i < 16 * T_ / 4; i += 32)
        S4[i] = make_float4(-1e30f, -1e30f, -1e30f, -1e30f);

    v16bf aq0 = load_a16(Q, HD_, q0, 0);
    v16bf aq1 = load_a16(Q, HD_, q0, 32);

    const int hl = lane >> 4, cc = lane & 15;
    const v8f vzero = {0.f,0.f,0.f,0.f,0.f,0.f,0.f,0.f};

    // S = (Q K^T) * HD^-0.5 over causal key tiles j <= qb
    for (int j = 0; j <= qb; ++j) {
        v8f acc = vzero;
        v16bf bk0 = load_b16(K, HD_, j * 16, 0);
        acc = wmma_bf16(aq0, bk0, acc);
        v16bf bk1 = load_b16(K, HD_, j * 16, 32);
        acc = wmma_bf16(aq1, bk1, acc);
        const int cg = j * 16 + cc;
#pragma unroll
        for (int r = 0; r < 8; ++r)
            S[r + hl * 8][cg] = acc[r] * 0.125f;   // 1/sqrt(64)
    }

    // causal mask inside the diagonal tile
    for (int i = lane; i < 256; i += 32) {
        int rr = i >> 4, c2 = i & 15;
        if (c2 > rr) S[rr][q0 + c2] = -1e30f;
    }

    // softmax: 2 lanes per row
    const int row = lane >> 1, sh = lane & 1;
    float mx = -1e30f;
    for (int c2 = sh * 128; c2 < sh * 128 + 128; ++c2)
        mx = fmaxf(mx, S[row][c2]);
    mx = fmaxf(mx, __shfl_xor(mx, 1, 32));
    float sum = 0.0f;
    for (int c2 = sh * 128; c2 < sh * 128 + 128; ++c2) {
        float e = __expf(S[row][c2] - mx);
        sum += e;
        P[row][c2] = f2bf(e);
    }
    sum += __shfl_xor(sum, 1, 32);
    if (sh == 0) rinv[row] = 1.0f / sum;

    // O = P @ V  (K-dim = valid keys, rounded up to 32; padded P is exp(-inf)=0)
    v8f ao[4];
#pragma unroll
    for (int ni = 0; ni < 4; ++ni) ao[ni] = vzero;
    const unsigned short* Pp = &P[0][0];
    const int nsteps = (qb + 2) >> 1;          // ceil((qb+1)*16 / 32)
    for (int jj = 0; jj < nsteps; ++jj) {
        const int k0 = jj * 32;
        v16bf ap = load_a16(Pp, T_, 0, k0);
#pragma unroll
        for (int ni = 0; ni < 4; ++ni) {
            v16bf bv = load_b16(V, T_, ni * 16, k0);
            ao[ni] = wmma_bf16(ap, bv, ao[ni]);
        }
    }

    // scale by 1/rowsum and store bf16 to [B,T,C] at column h*64+d
#pragma unroll
    for (int ni = 0; ni < 4; ++ni)
#pragma unroll
        for (int r = 0; r < 8; ++r) {
            int m = r + hl * 8;
            float val = ao[ni][r] * rinv[m];
            size_t o = ((size_t)(b * T_ + q0 + m)) * C_ + h * HD_ + ni * 16 + cc;
            Ob[o] = f2bf(val);
        }
}

// ------------------------------------------------------- output projection --

__global__ __launch_bounds__(32) void proj_gemm_kernel(
        const unsigned short* __restrict__ X,   // Ob bf16 [M_,C_]
        const unsigned short* __restrict__ Wp,  // bf16 [C_,C_]
        const float* __restrict__ bp,
        float* __restrict__ out) {
    const int m0 = blockIdx.x * 32;
    const int n0 = blockIdx.y * 64;

    const v8f vzero = {0.f,0.f,0.f,0.f,0.f,0.f,0.f,0.f};
    v8f acc[2][4];
#pragma unroll
    for (int i = 0; i < 2; ++i)
#pragma unroll
        for (int j = 0; j < 4; ++j) acc[i][j] = vzero;

    for (int k0 = 0; k0 < C_; k0 += 32) {
        v16bf a0 = load_a16(X, C_, m0, k0);
        v16bf a1 = load_a16(X, C_, m0 + 16, k0);
#pragma unroll
        for (int ni = 0; ni < 4; ++ni) {
            v16bf b = load_b16(Wp, C_, n0 + ni * 16, k0);
            acc[0][ni] = wmma_bf16(a0, b, acc[0][ni]);
            acc[1][ni] = wmma_bf16(a1, b, acc[1][ni]);
        }
    }

    const int lane = threadIdx.x & 31;
    const int hl = lane >> 4, c = lane & 15;
    float bias[4];
#pragma unroll
    for (int ni = 0; ni < 4; ++ni) bias[ni] = bp[n0 + ni * 16 + c];

#pragma unroll
    for (int mi = 0; mi < 2; ++mi)
#pragma unroll
        for (int ni = 0; ni < 4; ++ni)
#pragma unroll
            for (int r = 0; r < 8; ++r) {
                int m = m0 + mi * 16 + r + hl * 8;
                out[(size_t)m * C_ + n0 + ni * 16 + c] = acc[mi][ni][r] + bias[ni];
            }
}

// ------------------------------------------------------------------ launch --

extern "C" void kernel_launch(void* const* d_in, const int* in_sizes, int n_in,
                              void* d_out, int out_size, void* d_ws, size_t ws_size,
                              hipStream_t stream) {
    (void)in_sizes; (void)n_in; (void)out_size; (void)ws_size;

    const float* x  = (const float*)d_in[0];
    const float* Wq = (const float*)d_in[1];
    const float* Wk = (const float*)d_in[2];
    const float* Wv = (const float*)d_in[3];
    const float* Wp = (const float*)d_in[4];
    const float* bp = (const float*)d_in[5];
    float* out = (float*)d_out;

    const size_t nX = (size_t)M_ * C_;   // 25,165,824
    const size_t nW = (size_t)C_ * C_;   // 147,456

    char* p = (char*)d_ws;
    unsigned short* xb  = (unsigned short*)p; p += nX * 2;
    unsigned short* wqb = (unsigned short*)p; p += nW * 2;
    unsigned short* wkb = (unsigned short*)p; p += nW * 2;
    unsigned short* wvb = (unsigned short*)p; p += nW * 2;
    unsigned short* wpb = (unsigned short*)p; p += nW * 2;
    unsigned short* Qh  = (unsigned short*)p; p += nX * 2;
    unsigned short* Kh  = (unsigned short*)p; p += nX * 2;
    unsigned short* Vh  = (unsigned short*)p; p += nX * 2;
    unsigned short* Vt  = (unsigned short*)p; p += nX * 2;
    unsigned short* Ob  = (unsigned short*)p; p += nX * 2;

    f32_to_bf16_kernel<<<dim3(2048), dim3(256), 0, stream>>>(x, xb, (int)nX);
    f32_to_bf16_kernel<<<dim3(72), dim3(256), 0, stream>>>(Wq, wqb, (int)nW);
    f32_to_bf16_kernel<<<dim3(72), dim3(256), 0, stream>>>(Wk, wkb, (int)nW);
    f32_to_bf16_kernel<<<dim3(72), dim3(256), 0, stream>>>(Wv, wvb, (int)nW);
    f32_to_bf16_kernel<<<dim3(72), dim3(256), 0, stream>>>(Wp, wpb, (int)nW);

    qkv_gemm_kernel<<<dim3(M_ / 32, H_, 3), dim3(32), 0, stream>>>(
        xb, wqb, wkb, wvb, Qh, Kh, Vh);

    vtrans_kernel<<<dim3(B_ * H_), dim3(256), 0, stream>>>(Vh, Vt);

    attn_kernel<<<dim3(T_ / 16, H_, B_), dim3(32), 0, stream>>>(Qh, Kh, Vt, Ob);

    proj_gemm_kernel<<<dim3(M_ / 32, C_ / 64), dim3(32), 0, stream>>>(
        Ob, wpb, bp, out);
}
